// EncoderBlock_9526237462904
// MI455X (gfx1250) — matrix operand order repaired and checked by reading, not compiled, not run on hardware
//
#include <hip/hip_runtime.h>
#include <hip/hip_bf16.h>

// ---------------- CDNA5 WMMA types ----------------
typedef __attribute__((ext_vector_type(16))) __bf16 v16bf;
typedef __attribute__((ext_vector_type(8)))  __bf16 v8bf;
typedef __attribute__((ext_vector_type(8)))  float  v8f;

// ---------------- optional CDNA5 async VMEM->LDS path ----------------
#if defined(__gfx1250__) && defined(__has_builtin)
#if __has_builtin(__builtin_amdgcn_global_load_async_to_lds_b128) && \
    __has_builtin(__builtin_amdgcn_s_wait_asynccnt)
#define USE_ASYNC_LDS 1
#endif
#endif
#ifndef USE_ASYNC_LDS
#define USE_ASYNC_LDS 0
#endif

#if USE_ASYNC_LDS
typedef int v4i_pay __attribute__((vector_size(16)));
typedef __attribute__((address_space(1))) v4i_pay as1_v4i;   // global-side payload
typedef __attribute__((address_space(3))) v4i_pay as3_v4i;   // LDS-side payload
static __device__ __forceinline__ void async_copy16(const void* g, void* l) {
    __builtin_amdgcn_global_load_async_to_lds_b128((as1_v4i*)g, (as3_v4i*)l, 0, 0);
}
static __device__ __forceinline__ void async_wait0() {
    __builtin_amdgcn_s_wait_asynccnt(0);
}
#endif

// ---------------- problem constants ----------------
constexpr int Bb = 8;
constexpr int Nn = 1024;
constexpr int Dd = 768;
constexpr int Hh = 12;
constexpr int HD = 64;          // head dim
constexpr int Ff = 3072;
constexpr int Ss = Bb * Nn;     // 8192 rows
constexpr int HK = Hh * HD;     // 768
constexpr float EPSV = 1e-5f;

// ---------------- helpers ----------------
static __device__ __forceinline__ __bf16 f2bf(float f) {
    unsigned u = __builtin_bit_cast(unsigned, f);
    u += 0x7fffu + ((u >> 16) & 1u);          // round-to-nearest-even
    unsigned short hs = (unsigned short)(u >> 16);
    return __builtin_bit_cast(__bf16, hs);
}

static __device__ __forceinline__ v16bf cat16(v8bf lo, v8bf hi) {
    return __builtin_shufflevector(lo, hi, 0,1,2,3,4,5,6,7,8,9,10,11,12,13,14,15);
}

static __device__ __forceinline__ float gelu_exact(float x) {
    return 0.5f * x * (1.0f + erff(x * 0.7071067811865476f));
}

// ---------------- f32 -> bf16 convert ----------------
__global__ void f32_to_bf16_kernel(const float* __restrict__ in,
                                   __bf16* __restrict__ out, int n) {
    for (int i = blockIdx.x * blockDim.x + threadIdx.x; i < n;
         i += gridDim.x * blockDim.x)
        out[i] = f2bf(in[i]);
}

// ---------------- LayerNorm (f32 in) -> bf16 out ----------------
__global__ __launch_bounds__(256)
void ln_bf16_kernel(const float* __restrict__ x, const float* __restrict__ g,
                    const float* __restrict__ b, __bf16* __restrict__ out, int Dp) {
    const int row = blockIdx.x;
    const float* xr = x + (size_t)row * Dp;
    __shared__ float red[256];
    const int tid = threadIdx.x;

    float s = 0.f;
    for (int i = tid; i < Dp; i += 256) s += xr[i];
    red[tid] = s; __syncthreads();
    for (int off = 128; off > 0; off >>= 1) {
        if (tid < off) red[tid] += red[tid + off];
        __syncthreads();
    }
    const float mean = red[0] / (float)Dp;
    __syncthreads();

    float vs = 0.f;
    for (int i = tid; i < Dp; i += 256) { float d = xr[i] - mean; vs += d * d; }
    red[tid] = vs; __syncthreads();
    for (int off = 128; off > 0; off >>= 1) {
        if (tid < off) red[tid] += red[tid + off];
        __syncthreads();
    }
    const float rstd = rsqrtf(red[0] / (float)Dp + EPSV);

    for (int i = tid; i < Dp; i += 256)
        out[(size_t)row * Dp + i] = f2bf((xr[i] - mean) * rstd * g[i] + b[i]);
}

// ---------------- tiled bf16 WMMA GEMM ----------------
// C[M,Np] = A[M,Kd] * Bw[Kd,Np] + bias[Np]  (+ epilogue)
// MODE 0: bf16 out    MODE 1: bf16 out with exact GELU    MODE 2: f32 out + resid
template <int MODE>
__global__ __launch_bounds__(256)
void gemm_bf16_kernel(const __bf16* __restrict__ A, const __bf16* __restrict__ Bw,
                      const float* __restrict__ bias, const float* __restrict__ resid,
                      void* __restrict__ Cout, int M, int Np, int Kd) {
    constexpr int SA = 40;   // padded LDS stride (80B, 16B aligned)
    __shared__ __align__(16) __bf16 As[128 * SA];
    __shared__ __align__(16) __bf16 Bs[128 * SA];

    const int N0 = blockIdx.x * 128;
    const int M0 = blockIdx.y * 128;
    const int tid = threadIdx.x;
    const int lane = tid & 31;
    const int w = tid >> 5;
    const int waveM = w & 3;   // 4 waves along M (32 rows each)
    const int waveN = w >> 2;  // 2 waves along N (64 cols each)
    const int lg = lane >> 4;  // 16-lane group
    const int ll = lane & 15;

    v8f acc[2][4];
    #pragma unroll
    for (int mi = 0; mi < 2; mi++)
        #pragma unroll
        for (int ni = 0; ni < 4; ni++) acc[mi][ni] = v8f{};

    for (int k0 = 0; k0 < Kd; k0 += 32) {
        __syncthreads();
        // ---- stage A tile (128x32) row-major ----
        {
            const int r = tid >> 1, half = tid & 1;
            const __bf16* src = A + (size_t)(M0 + r) * Kd + k0 + half * 16;
#if USE_ASYNC_LDS
            // CDNA5 async VMEM->LDS: bypass VGPRs, tracked by ASYNCcnt
            async_copy16(src,     &As[r * SA + half * 16]);
            async_copy16(src + 8, &As[r * SA + half * 16 + 8]);
#else
            const uint4* s4 = (const uint4*)src;
            uint4 d0 = s4[0], d1 = s4[1];
            *(uint4*)&As[r * SA + half * 16]     = d0;
            *(uint4*)&As[r * SA + half * 16 + 8] = d1;
#endif
        }
        // ---- stage B tile (32x128) transposed -> Bs[n][k] ----
        #pragma unroll
        for (int i = 0; i < 2; i++) {
            const int c = tid + i * 256;            // 512 chunks of 8
            const int kk = c >> 4;
            const int nn0 = (c & 15) * 8;
            uint4 d = *(const uint4*)(Bw + (size_t)(k0 + kk) * Np + N0 + nn0);
            const __bf16* e = (const __bf16*)&d;
            #pragma unroll
            for (int j = 0; j < 8; j++) Bs[(nn0 + j) * SA + kk] = e[j];
        }
        if (k0 + 32 < Kd)
            __builtin_prefetch(A + (size_t)(M0 + (tid >> 1)) * Kd + k0 + 32, 0, 3);
#if USE_ASYNC_LDS
        async_wait0();
#endif
        __syncthreads();

        // ---- fragments + WMMA ----
        v16bf af[2];
        #pragma unroll
        for (int mi = 0; mi < 2; mi++) {
            const int row = waveM * 32 + mi * 16 + ll;
            const int kb = lg * 8;
            v8bf lo = *(const v8bf*)&As[row * SA + kb];
            v8bf hi = *(const v8bf*)&As[row * SA + 16 + kb];
            af[mi] = cat16(lo, hi);
        }
        #pragma unroll
        for (int ni = 0; ni < 4; ni++) {
            const int coln = waveN * 64 + ni * 16 + ll;
            const int ko = lg * 8;
            v8bf lo = *(const v8bf*)&Bs[coln * SA + ko];
            v8bf hi = *(const v8bf*)&Bs[coln * SA + ko + 16];
            v16bf bf = cat16(lo, hi);
            #pragma unroll
            for (int mi = 0; mi < 2; mi++)
                acc[mi][ni] = __builtin_amdgcn_wmma_f32_16x16x32_bf16(
                    false, af[mi], false, bf, (short)0, acc[mi][ni], false, false);
        }
    }

    // ---- epilogue (C layout: vgpr v -> M = v + lg*8, N = ll) ----
    #pragma unroll
    for (int mi = 0; mi < 2; mi++)
        #pragma unroll
        for (int ni = 0; ni < 4; ni++)
            #pragma unroll
            for (int v = 0; v < 8; v++) {
                const int row = M0 + waveM * 32 + mi * 16 + v + lg * 8;
                const int coln = N0 + waveN * 64 + ni * 16 + ll;
                float val = acc[mi][ni][v] + bias[coln];
                if (MODE == 1) val = gelu_exact(val);
                if (MODE == 2) {
                    float* Cf = (float*)Cout;
                    Cf[(size_t)row * Np + coln] = val + resid[(size_t)row * Np + coln];
                } else {
                    __bf16* Cb = (__bf16*)Cout;
                    Cb[(size_t)row * Np + coln] = f2bf(val);
                }
            }
}

// ---------------- flash attention (per b,h, 64 query rows / block) ----------------
__global__ __launch_bounds__(128)
void attn_kernel(const __bf16* __restrict__ q, const __bf16* __restrict__ k,
                 const __bf16* __restrict__ v, __bf16* __restrict__ ctx_out,
                 float scale) {
    constexpr int SP = 72;   // padded stride (144B, 16B aligned)
    __shared__ __align__(16) __bf16 Qs[64 * SP];
    __shared__ __align__(16) __bf16 Ks[64 * SP];
    __shared__ __align__(16) __bf16 Vs[64 * SP];   // transposed: [dim][key]
    __shared__ __align__(16) __bf16 Ps[64 * SP];

    const int q0 = blockIdx.x * 64;
    const int h  = blockIdx.y;
    const int b  = blockIdx.z;
    const int tid = threadIdx.x;
    const int lane = tid & 31;
    const int w = tid >> 5;      // 4 waves, 16 query rows each
    const int lg = lane >> 4;
    const int ll = lane & 15;

    // ---- stage Q tile [64 rows x 64 dims] ----
    #pragma unroll
    for (int i = 0; i < 4; i++) {
        const int c = tid + i * 128;           // 512 chunks of 8
        const int r = c >> 3, cc = (c & 7) * 8;
        const __bf16* src = q + (size_t)(b * Nn + q0 + r) * HK + h * HD + cc;
#if USE_ASYNC_LDS
        async_copy16(src, &Qs[r * SP + cc]);
#else
        *(uint4*)&Qs[r * SP + cc] = *(const uint4*)src;
#endif
    }
#if USE_ASYNC_LDS
    async_wait0();
#endif
    __syncthreads();

    // Q A-fragments (two 32-wide k-steps over head dim)
    v16bf qf[2];
    #pragma unroll
    for (int ks = 0; ks < 2; ks++) {
        const int row = w * 16 + ll;
        const int kb = ks * 32 + lg * 8;
        v8bf lo = *(const v8bf*)&Qs[row * SP + kb];
        v8bf hi = *(const v8bf*)&Qs[row * SP + kb + 16];
        qf[ks] = cat16(lo, hi);
    }

    float mrow[8], lrow[8];
    v8f cacc[4];
    #pragma unroll
    for (int i = 0; i < 8; i++) { mrow[i] = -1e30f; lrow[i] = 0.f; }
    #pragma unroll
    for (int ni = 0; ni < 4; ni++) cacc[ni] = v8f{};

    for (int kb0 = 0; kb0 < Nn; kb0 += 64) {
        __syncthreads();
        // ---- stage K tile (natural, async-capable) + V tile (transposed) ----
        #pragma unroll
        for (int i = 0; i < 4; i++) {
            const int c = tid + i * 128;
            const int r = c >> 3, cc = (c & 7) * 8;
            const __bf16* ksrc = k + (size_t)(b * Nn + kb0 + r) * HK + h * HD + cc;
#if USE_ASYNC_LDS
            async_copy16(ksrc, &Ks[r * SP + cc]);
#else
            *(uint4*)&Ks[r * SP + cc] = *(const uint4*)ksrc;
#endif
            uint4 vd = *(const uint4*)(v + (size_t)(b * Nn + kb0 + r) * HK + h * HD + cc);
            const __bf16* e = (const __bf16*)&vd;
            #pragma unroll
            for (int j = 0; j < 8; j++) Vs[(cc + j) * SP + r] = e[j];
        }
#if USE_ASYNC_LDS
        async_wait0();
#endif
        __syncthreads();

        // ---- scores = Q K^T ----
        v8f s[4];
        #pragma unroll
        for (int ni = 0; ni < 4; ni++) s[ni] = v8f{};
        #pragma unroll
        for (int ks = 0; ks < 2; ks++)
            #pragma unroll
            for (int ni = 0; ni < 4; ni++) {
                const int coln = ni * 16 + ll;                 // key index
                const int ko = ks * 32 + lg * 8;               // head-dim offset
                v8bf lo = *(const v8bf*)&Ks[coln * SP + ko];
                v8bf hi = *(const v8bf*)&Ks[coln * SP + ko + 16];
                s[ni] = __builtin_amdgcn_wmma_f32_16x16x32_bf16(
                    false, qf[ks], false, cat16(lo, hi), (short)0, s[ni], false, false);
            }

        // ---- online softmax (row = v + lg*8 within wave tile) ----
        #pragma unroll
        for (int vv = 0; vv < 8; vv++) {
            float lm = -1e30f;
            #pragma unroll
            for (int ni = 0; ni < 4; ni++) {
                float t = s[ni][vv] * scale;
                s[ni][vv] = t;
                lm = fmaxf(lm, t);
            }
            #pragma unroll
            for (int off = 1; off < 16; off <<= 1)
                lm = fmaxf(lm, __shfl_xor(lm, off, 32));
            const float nm = fmaxf(mrow[vv], lm);
            const float alpha = __expf(mrow[vv] - nm);
            float lsum = 0.f;
            #pragma unroll
            for (int ni = 0; ni < 4; ni++) {
                float p = __expf(s[ni][vv] - nm);
                s[ni][vv] = p;
                lsum += p;
            }
            #pragma unroll
            for (int off = 1; off < 16; off <<= 1)
                lsum += __shfl_xor(lsum, off, 32);
            lrow[vv] = lrow[vv] * alpha + lsum;
            mrow[vv] = nm;
            #pragma unroll
            for (int ni = 0; ni < 4; ni++) cacc[ni][vv] *= alpha;
            const int prow = w * 16 + vv + lg * 8;
            #pragma unroll
            for (int ni = 0; ni < 4; ni++)
                Ps[prow * SP + ni * 16 + ll] = f2bf(s[ni][vv]);
        }
        __syncthreads();

        // ---- ctx += P @ V ----
        #pragma unroll
        for (int ks = 0; ks < 2; ks++) {
            const int prw = w * 16 + ll;
            const int kb = ks * 32 + lg * 8;
            v8bf plo = *(const v8bf*)&Ps[prw * SP + kb];
            v8bf phi = *(const v8bf*)&Ps[prw * SP + kb + 16];
            v16bf pf = cat16(plo, phi);
            #pragma unroll
            for (int ni = 0; ni < 4; ni++) {
                const int coln = ni * 16 + ll;                 // output head-dim
                const int ko = ks * 32 + lg * 8;               // key offset
                v8bf lo = *(const v8bf*)&Vs[coln * SP + ko];
                v8bf hi = *(const v8bf*)&Vs[coln * SP + ko + 16];
                cacc[ni] = __builtin_amdgcn_wmma_f32_16x16x32_bf16(
                    false, pf, false, cat16(lo, hi), (short)0, cacc[ni], false, false);
            }
        }
    }

    // ---- normalize + write ctx [B,N,H,K] bf16 ----
    #pragma unroll
    for (int ni = 0; ni < 4; ni++)
        #pragma unroll
        for (int vv = 0; vv < 8; vv++) {
            const int qr = q0 + w * 16 + vv + lg * 8;
            const int dd = ni * 16 + ll;
            const float val = cacc[ni][vv] / lrow[vv];
            ctx_out[(size_t)(b * Nn + qr) * HK + h * HD + dd] = f2bf(val);
        }
}

// ---------------- host launch ----------------
extern "C" void kernel_launch(void* const* d_in, const int* in_sizes, int n_in,
                              void* d_out, int out_size, void* d_ws, size_t ws_size,
                              hipStream_t stream) {
    const float* x     = (const float*)d_in[0];
    const float* ln1_g = (const float*)d_in[1];
    const float* ln1_b = (const float*)d_in[2];
    const float* wq    = (const float*)d_in[3];
    const float* bq    = (const float*)d_in[4];
    const float* wk    = (const float*)d_in[5];
    const float* bk    = (const float*)d_in[6];
    const float* wv    = (const float*)d_in[7];
    const float* bv    = (const float*)d_in[8];
    const float* wo    = (const float*)d_in[9];
    const float* bo    = (const float*)d_in[10];
    const float* ln2_g = (const float*)d_in[11];
    const float* ln2_b = (const float*)d_in[12];
    const float* w1    = (const float*)d_in[13];
    const float* b1    = (const float*)d_in[14];
    const float* w2    = (const float*)d_in[15];
    const float* b2    = (const float*)d_in[16];

    char* p = (char*)d_ws;
    auto carve = [&](size_t bytes) -> void* {
        void* r = (void*)p;
        p += (bytes + 255) & ~(size_t)255;
        return r;
    };
    __bf16* wq_bf = (__bf16*)carve((size_t)Dd * HK * 2);
    __bf16* wk_bf = (__bf16*)carve((size_t)Dd * HK * 2);
    __bf16* wv_bf = (__bf16*)carve((size_t)Dd * HK * 2);
    __bf16* wo_bf = (__bf16*)carve((size_t)HK * Dd * 2);
    __bf16* w1_bf = (__bf16*)carve((size_t)Dd * Ff * 2);
    __bf16* w2_bf = (__bf16*)carve((size_t)Ff * Dd * 2);
    __bf16* h1    = (__bf16*)carve((size_t)Ss * Dd * 2);
    __bf16* qb    = (__bf16*)carve((size_t)Ss * HK * 2);
    __bf16* kb    = (__bf16*)carve((size_t)Ss * HK * 2);
    __bf16* vb    = (__bf16*)carve((size_t)Ss * HK * 2);
    __bf16* ctxb  = (__bf16*)carve((size_t)Ss * HK * 2);
    float*  ares  = (float*) carve((size_t)Ss * Dd * 4);
    __bf16* h2    = (__bf16*)carve((size_t)Ss * Dd * 2);
    __bf16* m1    = (__bf16*)carve((size_t)Ss * Ff * 2);

    // weight conversion
    f32_to_bf16_kernel<<<512, 256, 0, stream>>>(wq, wq_bf, Dd * HK);
    f32_to_bf16_kernel<<<512, 256, 0, stream>>>(wk, wk_bf, Dd * HK);
    f32_to_bf16_kernel<<<512, 256, 0, stream>>>(wv, wv_bf, Dd * HK);
    f32_to_bf16_kernel<<<512, 256, 0, stream>>>(wo, wo_bf, HK * Dd);
    f32_to_bf16_kernel<<<1024, 256, 0, stream>>>(w1, w1_bf, Dd * Ff);
    f32_to_bf16_kernel<<<1024, 256, 0, stream>>>(w2, w2_bf, Ff * Dd);

    // LN1
    ln_bf16_kernel<<<Ss, 256, 0, stream>>>(x, ln1_g, ln1_b, h1, Dd);

    // QKV projections
    dim3 gqkv(HK / 128, Ss / 128);
    gemm_bf16_kernel<0><<<gqkv, 256, 0, stream>>>(h1, wq_bf, bq, nullptr, qb, Ss, HK, Dd);
    gemm_bf16_kernel<0><<<gqkv, 256, 0, stream>>>(h1, wk_bf, bk, nullptr, kb, Ss, HK, Dd);
    gemm_bf16_kernel<0><<<gqkv, 256, 0, stream>>>(h1, wv_bf, bv, nullptr, vb, Ss, HK, Dd);

    // attention
    attn_kernel<<<dim3(Nn / 64, Hh, Bb), 128, 0, stream>>>(qb, kb, vb, ctxb, 0.125f);

    // output projection + residual (f32)
    gemm_bf16_kernel<2><<<dim3(Dd / 128, Ss / 128), 256, 0, stream>>>(
        ctxb, wo_bf, bo, x, ares, Ss, Dd, HK);

    // LN2
    ln_bf16_kernel<<<Ss, 256, 0, stream>>>(ares, ln2_g, ln2_b, h2, Dd);

    // MLP
    gemm_bf16_kernel<1><<<dim3(Ff / 128, Ss / 128), 256, 0, stream>>>(
        h2, w1_bf, b1, nullptr, m1, Ss, Ff, Dd);
    gemm_bf16_kernel<2><<<dim3(Dd / 128, Ss / 128), 256, 0, stream>>>(
        m1, w2_bf, b2, ares, (float*)d_out, Ss, Dd, Ff);
}
